// Model_25718264168479
// MI455X (gfx1250) — matrix operand order repaired and checked by reading, not compiled, not run on hardware
//
#include <hip/hip_runtime.h>
#include <math.h>

// ---- problem constants (match reference) ----
constexpr int V_    = 50000;
constexpr int E_    = 128;
constexpr int H_    = 256;
constexpr int H2_   = 512;
constexpr int B_    = 8;
constexpr int S_    = 400;
constexpr int T_    = 60;
constexpr int HS3_  = 3 * H2_;   // 1536

typedef __attribute__((ext_vector_type(16))) _Float16 v16h;
typedef __attribute__((ext_vector_type(8)))  float    v8f;

__device__ __forceinline__ float sigf(float x) { return 1.f / (1.f + __expf(-x)); }

// ---------------------------------------------------------------------------
// Embedding gather: emb[row, :] = embedding[input[row], :]   row = b*S + s
// ---------------------------------------------------------------------------
__global__ void embed_gather(const int* __restrict__ input,
                             const float* __restrict__ embedding,
                             float* __restrict__ emb) {
    int row = blockIdx.x;                 // B*S rows
    int tid = threadIdx.x;                // E threads
    int tok = input[row];
    emb[(size_t)row * E_ + tid] = embedding[(size_t)tok * E_ + tid];
}

// ---------------------------------------------------------------------------
// pre_d[col] = embedding[start_id] @ Wih_d[:,col] + b_d[col]   (2048 cols)
// ---------------------------------------------------------------------------
__global__ __launch_bounds__(256) void precompute_dec_in(
    const int* __restrict__ start_ptr, const float* __restrict__ embedding,
    const float* __restrict__ Wih_d, const float* __restrict__ b_d,
    float* __restrict__ pre) {
    int col = blockIdx.x * 256 + threadIdx.x;
    if (col >= 4 * H2_) return;
    const float* ev = embedding + (size_t)start_ptr[0] * E_;
    float acc = b_d[col];
    for (int k = 0; k < E_; ++k) acc += ev[k] * Wih_d[(size_t)k * (4 * H2_) + col];
    pre[col] = acc;
}

// ---------------------------------------------------------------------------
// GEMM  C(MxN) = A(MxK) @ B(KxN) + bias, fp32 in/out, f16 WMMA math.
// REQUIRES: M % 16 == 0, K % 32 == 0 (true for all call sites).
// Block = 256 threads = 8 waves; block tile = 16(M) x 256(N); K chunk = 32.
// Each wave owns two 16x16 output tiles sharing one A fragment.
//
// LDS tiles are stored in *fragment order* so fragment builds are wide loads:
//   A frag (f16 16x32, contiguous k = 16*hi + j): lane = m + 16*((k>>4)&1),
//                                   j    = k & 15
//     -> As[lane*16 + j], each lane reads 32B contiguous (ds_load_b128 x2)
//   B frag (f16 32x16): lane = n(+16*((k>>4)&1)), j = k & 15
//     -> Bs[col*40 + k] column-major, 80B col stride (16B-aligned, ~conflict
//        free on 64 banks); each lane reads 32B contiguous per tile.
//   C/D f32 16x16: N = lane&15, M = r + 8*(lane>>4)
// ---------------------------------------------------------------------------
__global__ __launch_bounds__(256) void gemm_wmma_f16(
    const float* __restrict__ A, int lda,
    const float* __restrict__ Bw, int ldb,
    const float* __restrict__ bias,
    float* __restrict__ C, int ldc,
    int M, int N, int K) {
    __shared__ alignas(16) _Float16 As[32 * 16];        // 1 KB, fragment order
    __shared__ alignas(16) _Float16 Bs[256 * 40];       // 20 KB, col-major padded

    int tid   = threadIdx.x;
    int nBase = blockIdx.x * 256;
    int mBase = blockIdx.y * 16;
    int lane = tid & 31, wv = tid >> 5;
    int laneLow = lane & 15, hi = lane >> 4;
    bool fullN = (nBase + 256 <= N);

    v8f acc0 = {}, acc1 = {};
    for (int k0 = 0; k0 < K; k0 += 32) {
        // ---- stage A (16x32) as float4 loads, write in fragment order ----
        if (tid < 128) {
            int m = tid >> 3, k4 = (tid & 7) << 2;
            const float4 f = *(const float4*)&A[(size_t)(mBase + m) * lda + k0 + k4];
            float fv[4] = {f.x, f.y, f.z, f.w};
#pragma unroll
            for (int c = 0; c < 4; ++c) {
                int k = k4 + c;
                int ln = m + (((k >> 4) & 1) << 4);
                int j  = k & 15;
                As[ln * 16 + j] = (_Float16)fv[c];
            }
        }
        // ---- stage B (32x256) ----
        if (fullN) {
            for (int i = tid; i < 2048; i += 256) {       // 2048 float4s
                int k = i >> 6, n4 = (i & 63) << 2;
                const float4 f = *(const float4*)&Bw[(size_t)(k0 + k) * ldb + nBase + n4];
                Bs[(n4 + 0) * 40 + k] = (_Float16)f.x;
                Bs[(n4 + 1) * 40 + k] = (_Float16)f.y;
                Bs[(n4 + 2) * 40 + k] = (_Float16)f.z;
                Bs[(n4 + 3) * 40 + k] = (_Float16)f.w;
            }
        } else {                                          // N-edge block only
            for (int i = tid; i < 8192; i += 256) {
                int k = i >> 8, n = i & 255;
                int gn = nBase + n;
                float v = (gn < N) ? Bw[(size_t)(k0 + k) * ldb + gn] : 0.f;
                Bs[n * 40 + k] = (_Float16)v;
            }
        }
        // stream-prefetch the next K-slice of B (global_prefetch_b8)
        {
            int pk = k0 + 32 + (tid >> 3);
            int pn = nBase + ((tid & 7) << 5);
            if (pk < K && pn < N)
                __builtin_prefetch(&Bw[(size_t)pk * ldb + pn], 0, 1);
        }
        __syncthreads();

        v16h a  = *(const v16h*)&As[lane * 16];
        v16h b0 = *(const v16h*)&Bs[(wv * 32 + laneLow) * 40 + hi * 16];
        v16h b1 = *(const v16h*)&Bs[(wv * 32 + 16 + laneLow) * 40 + hi * 16];
        acc0 = __builtin_amdgcn_wmma_f32_16x16x32_f16(false, a, false, b0,
                                                      (short)0, acc0, false, false);
        acc1 = __builtin_amdgcn_wmma_f32_16x16x32_f16(false, a, false, b1,
                                                      (short)0, acc1, false, false);
        __syncthreads();
    }
#pragma unroll
    for (int r = 0; r < 8; ++r) {
        int m  = mBase + r + (hi << 3);
        int n0 = nBase + (wv << 5) + laneLow;
        int n1 = n0 + 16;
        if (n0 < N) C[(size_t)m * ldc + n0] = acc0[r] + bias[n0];
        if (n1 < N) C[(size_t)m * ldc + n1] = acc1[r] + bias[n1];
    }
}

// ---------------------------------------------------------------------------
// Bidirectional encoder LSTM scan. One persistent 1024-thread block per
// direction (blockIdx.x). h lives in LDS; c lives in registers; X = emb@Wih+b
// precomputed. Each thread owns 2 (b, j) hidden units (2048 units total).
// ---------------------------------------------------------------------------
__global__ __launch_bounds__(1024) void lstm_encoder(
    const float* __restrict__ Xf, const float* __restrict__ Xb,
    const float* __restrict__ Whh_f, const float* __restrict__ Whh_b,
    float* __restrict__ h_enc, float* __restrict__ h0, float* __restrict__ c0) {
    int dir = blockIdx.x;
    const float* X = dir ? Xb : Xf;
    const float* W = dir ? Whh_b : Whh_f;
    __shared__ float h_s[B_ * H_];     // 8KB
    int tid = threadIdx.x;
    float c_reg[2] = {0.f, 0.f};
    for (int i = tid; i < B_ * H_; i += 1024) h_s[i] = 0.f;
    __syncthreads();

    for (int step = 0; step < S_; ++step) {
        int t = dir ? (S_ - 1 - step) : step;
        float nh[2];
#pragma unroll
        for (int r = 0; r < 2; ++r) {
            int u = tid + 1024 * r;
            int b = u >> 8, j = u & 255;
            const float* xrow = X + (size_t)(b * S_ + t) * (4 * H_);
            float zi = xrow[j], zf = xrow[H_ + j], zg = xrow[2 * H_ + j], zo = xrow[3 * H_ + j];
            const float* hb = h_s + b * H_;
            for (int k = 0; k < H_; ++k) {
                float hv = hb[k];
                const float* wr = W + (size_t)k * (4 * H_) + j;
                zi += hv * wr[0]; zf += hv * wr[H_]; zg += hv * wr[2 * H_]; zo += hv * wr[3 * H_];
            }
            float cn = sigf(zf) * c_reg[r] + sigf(zi) * tanhf(zg);
            c_reg[r] = cn;
            nh[r] = sigf(zo) * tanhf(cn);
            h_enc[(size_t)(b * S_ + t) * H2_ + dir * H_ + j] = nh[r];
        }
        __syncthreads();
#pragma unroll
        for (int r = 0; r < 2; ++r) h_s[tid + 1024 * r] = nh[r];  // u == b*H_+j
        __syncthreads();
    }
#pragma unroll
    for (int r = 0; r < 2; ++r) {
        int u = tid + 1024 * r;
        int b = u >> 8, j = u & 255;
        h0[b * H2_ + dir * H_ + j] = h_s[u];
        c0[b * H2_ + dir * H_ + j] = c_reg[r];
    }
}

// ---------------------------------------------------------------------------
// Decoder 60-step scan: LSTMCell + intra-temporal attention + intra-decoder
// attention + p_u. One persistent 1024-thread block; h in LDS, c in regs.
// ---------------------------------------------------------------------------
__global__ __launch_bounds__(1024) void decoder_scan(
    const int* __restrict__ input, const int* __restrict__ pad_ptr,
    const float* __restrict__ pre, const float* __restrict__ Whh_d,
    const float* __restrict__ W_e, const float* __restrict__ W_d,
    const float* __restrict__ h_enc, const float* __restrict__ h0,
    const float* __restrict__ c0, const float* __restrict__ lin_u_w,
    const float* __restrict__ lin_u_b,
    float* __restrict__ buf, float* __restrict__ hs_all,
    float* __restrict__ p_u, float* __restrict__ alphat,
    float* __restrict__ sum_g) {
    __shared__ float h_s[B_ * H2_];    // h_t        (16KB)
    __shared__ float tmpA[B_ * H2_];   // h_new / ct_e
    __shared__ float tmpB[B_ * H2_];   // q_e / q_d / ct_d
    __shared__ float ad[B_ * T_];      // decoder-attn weights
    __shared__ float red[B_];

    int tid = threadIdx.x;
    int pad_id = pad_ptr[0];
    float c_reg[4];
#pragma unroll
    for (int r = 0; r < 4; ++r) {
        int idx = tid + 1024 * r;
        h_s[idx] = h0[idx];
        c_reg[r] = c0[idx];
    }
    __syncthreads();

    for (int t = 0; t < T_; ++t) {
        // --- LSTMCell gates: z = pre + h @ Whh_d ---
#pragma unroll
        for (int r = 0; r < 4; ++r) {
            int idx = tid + 1024 * r;
            int b = idx >> 9, j = idx & 511;
            float zi = pre[j], zf = pre[H2_ + j], zg = pre[2 * H2_ + j], zo = pre[3 * H2_ + j];
            const float* hb = h_s + b * H2_;
            for (int k = 0; k < H2_; ++k) {
                float hv = hb[k];
                const float* wr = Whh_d + (size_t)k * (4 * H2_) + j;
                zi += hv * wr[0]; zf += hv * wr[H2_]; zg += hv * wr[2 * H2_]; zo += hv * wr[3 * H2_];
            }
            float cn = sigf(zf) * c_reg[r] + sigf(zi) * tanhf(zg);
            c_reg[r] = cn;
            float nh = sigf(zo) * tanhf(cn);
            tmpA[idx] = nh;
            buf[(size_t)(b * T_ + t) * H2_ + j] = nh;
            hs_all[(size_t)(t * B_ + b) * HS3_ + j] = nh;
        }
        __syncthreads();
#pragma unroll
        for (int r = 0; r < 4; ++r) { int idx = tid + 1024 * r; h_s[idx] = tmpA[idx]; }
        __syncthreads();

        // --- q_e = h_t @ W_e ---
#pragma unroll
        for (int r = 0; r < 4; ++r) {
            int idx = tid + 1024 * r;
            int b = idx >> 9, dc = idx & 511;
            float q = 0.f;
            const float* hb = h_s + b * H2_;
            for (int k = 0; k < H2_; ++k) q += hb[k] * W_e[(size_t)k * H2_ + dc];
            tmpB[idx] = q;
        }
        if (tid < B_) red[tid] = 0.f;
        __syncthreads();

        // --- scores over encoder, running exp-normalization, mask ---
        for (int idx = tid; idx < B_ * S_; idx += 1024) {
            int b = idx / S_;
            const float* qb = tmpB + b * H2_;
            const float* he = h_enc + (size_t)idx * H2_;
            float sc = 0.f;
            for (int d = 0; d < H2_; ++d) sc += qb[d] * he[d];
            float e = __expf(sc);
            float sold = sum_g[idx];
            float nrm = (t == 0) ? e : e / sold;
            sum_g[idx] = (t == 0) ? (1e-10f + e) : (sold + e);
            float m = (input[idx] != pad_id) ? 1.f : 0.f;
            float a = nrm * m;
            alphat[(size_t)t * (B_ * S_) + idx] = a;
            atomicAdd(&red[b], a);
        }
        __syncthreads();
        for (int idx = tid; idx < B_ * S_; idx += 1024) {
            int b = idx / S_;
            alphat[(size_t)t * (B_ * S_) + idx] /= red[b];
        }
        __syncthreads();

        // --- ct_e = alphat @ h_enc ---
#pragma unroll
        for (int r = 0; r < 4; ++r) {
            int idx = tid + 1024 * r;
            int b = idx >> 9, d = idx & 511;
            float acc = 0.f;
            const float* ar = alphat + (size_t)t * (B_ * S_) + b * S_;
            for (int s = 0; s < S_; ++s) acc += ar[s] * h_enc[(size_t)(b * S_ + s) * H2_ + d];
            tmpA[idx] = acc;
            hs_all[(size_t)(t * B_ + b) * HS3_ + H2_ + d] = acc;
        }
        __syncthreads();

        // --- q_d = h_t @ W_d ---
#pragma unroll
        for (int r = 0; r < 4; ++r) {
            int idx = tid + 1024 * r;
            int b = idx >> 9, dc = idx & 511;
            float q = 0.f;
            const float* hb = h_s + b * H2_;
            for (int k = 0; k < H2_; ++k) q += hb[k] * W_d[(size_t)k * H2_ + dc];
            tmpB[idx] = q;
        }
        __syncthreads();

        // --- intra-decoder attention over previous states ---
        for (int idx = tid; idx < B_ * T_; idx += 1024) {
            int b = idx / T_, tt = idx - b * T_;
            float sd = -1e9f;
            if (tt < t) {
                sd = 0.f;
                const float* qb = tmpB + b * H2_;
                const float* br = buf + (size_t)(b * T_ + tt) * H2_;
                for (int d = 0; d < H2_; ++d) sd += qb[d] * br[d];
            }
            ad[idx] = sd;
        }
        __syncthreads();
        if (tid < B_) {
            float mx = -3e38f;
            for (int tt = 0; tt < T_; ++tt) mx = fmaxf(mx, ad[tid * T_ + tt]);
            float sm = 0.f;
            for (int tt = 0; tt < T_; ++tt) { float e = __expf(ad[tid * T_ + tt] - mx); ad[tid * T_ + tt] = e; sm += e; }
            float inv = 1.f / sm;
            for (int tt = 0; tt < T_; ++tt) ad[tid * T_ + tt] *= inv;
        }
        __syncthreads();

        // --- ct_d = alpha_d @ buf (zero at t==0) ---
#pragma unroll
        for (int r = 0; r < 4; ++r) {
            int idx = tid + 1024 * r;
            int b = idx >> 9, d = idx & 511;
            float acc = 0.f;
            if (t > 0) {
                const float* ab = ad + b * T_;
                for (int tt = 0; tt < T_; ++tt) acc += ab[tt] * buf[(size_t)(b * T_ + tt) * H2_ + d];
            }
            tmpB[idx] = acc;
            hs_all[(size_t)(t * B_ + b) * HS3_ + 2 * H2_ + d] = acc;
        }
        if (tid < B_) red[tid] = 0.f;
        __syncthreads();

        // --- p_u = sigmoid(hs . lin_u_w + b), hs = [h_t, ct_e, ct_d] (all LDS) ---
        for (int n = tid; n < B_ * HS3_; n += 1024) {
            int b = n / HS3_, k = n - b * HS3_;
            float v = (k < H2_) ? h_s[b * H2_ + k]
                    : (k < 2 * H2_) ? tmpA[b * H2_ + (k - H2_)]
                                    : tmpB[b * H2_ + (k - 2 * H2_)];
            atomicAdd(&red[b], v * lin_u_w[k]);
        }
        __syncthreads();
        if (tid < B_) p_u[t * B_ + tid] = sigf(red[tid] + lin_u_b[0]);
        __syncthreads();
    }
}

// ---------------------------------------------------------------------------
// Row softmax over V, scaled by p_u[r]: out[r,:] = p_u[r]*softmax(out[r,:])
// ---------------------------------------------------------------------------
__global__ __launch_bounds__(256) void softmax_scale(float* __restrict__ out,
                                                     const float* __restrict__ p_u) {
    int r = blockIdx.x;
    float* row = out + (size_t)r * V_;
    __shared__ float sdata[256];
    int tid = threadIdx.x;
    float mx = -3e38f;
    for (int i = tid; i < V_; i += 256) mx = fmaxf(mx, row[i]);
    sdata[tid] = mx; __syncthreads();
    for (int o = 128; o > 0; o >>= 1) { if (tid < o) sdata[tid] = fmaxf(sdata[tid], sdata[tid + o]); __syncthreads(); }
    mx = sdata[0]; __syncthreads();
    float sm = 0.f;
    for (int i = tid; i < V_; i += 256) sm += __expf(row[i] - mx);
    sdata[tid] = sm; __syncthreads();
    for (int o = 128; o > 0; o >>= 1) { if (tid < o) sdata[tid] += sdata[tid + o]; __syncthreads(); }
    float scale = p_u[r] / sdata[0];
    for (int i = tid; i < V_; i += 256) row[i] = __expf(row[i] - mx) * scale;
}

// ---------------------------------------------------------------------------
// Pointer scatter: out[t,b,input[b,s]] += (1-p_u[t,b]) * alphat[t,b,s]
// ---------------------------------------------------------------------------
__global__ void scatter_ptr(float* __restrict__ out, const float* __restrict__ p_u,
                            const float* __restrict__ alphat,
                            const int* __restrict__ input) {
    int idx = blockIdx.x * blockDim.x + threadIdx.x;
    if (idx >= T_ * B_ * S_) return;
    int t = idx / (B_ * S_);
    int rem = idx - t * (B_ * S_);
    int b = rem / S_, s = rem - b * S_;
    float val = (1.f - p_u[t * B_ + b]) * alphat[idx];
    int tok = input[b * S_ + s];
    atomicAdd(&out[(size_t)(t * B_ + b) * V_ + tok], val);
}

// ---------------------------------------------------------------------------
extern "C" void kernel_launch(void* const* d_in, const int* in_sizes, int n_in,
                              void* d_out, int out_size, void* d_ws, size_t ws_size,
                              hipStream_t stream) {
    (void)in_sizes; (void)n_in; (void)out_size; (void)ws_size;
    const int*   input     = (const int*)d_in[0];
    const int*   start_id  = (const int*)d_in[1];
    const int*   pad_id    = (const int*)d_in[2];
    const float* embedding = (const float*)d_in[3];
    const float* Wih_f     = (const float*)d_in[4];
    const float* Whh_f     = (const float*)d_in[5];
    const float* b_f       = (const float*)d_in[6];
    const float* Wih_b     = (const float*)d_in[7];
    const float* Whh_b     = (const float*)d_in[8];
    const float* b_b       = (const float*)d_in[9];
    const float* Wih_d     = (const float*)d_in[10];
    const float* Whh_d     = (const float*)d_in[11];
    const float* b_d       = (const float*)d_in[12];
    const float* W_e       = (const float*)d_in[13];
    const float* W_d       = (const float*)d_in[14];
    const float* lin_out_w = (const float*)d_in[15];
    const float* lin_out_b = (const float*)d_in[16];
    const float* lin_u_w   = (const float*)d_in[17];
    const float* lin_u_b   = (const float*)d_in[18];
    float* out = (float*)d_out;

    // workspace carve-up (~39 MB of fp32)
    float* w = (float*)d_ws;
    float* emb    = w; w += (size_t)B_ * S_ * E_;        // 409600
    float* Xf     = w; w += (size_t)B_ * S_ * 4 * H_;    // 3276800
    float* Xb     = w; w += (size_t)B_ * S_ * 4 * H_;    // 3276800
    float* h_enc  = w; w += (size_t)B_ * S_ * H2_;       // 1638400
    float* h0     = w; w += (size_t)B_ * H2_;            // 4096
    float* c0     = w; w += (size_t)B_ * H2_;            // 4096
    float* pre    = w; w += (size_t)4 * H2_;             // 2048
    float* buf    = w; w += (size_t)B_ * T_ * H2_;       // 245760
    float* hs_all = w; w += (size_t)T_ * B_ * HS3_;      // 737280
    float* p_u    = w; w += (size_t)T_ * B_;             // 480
    float* alphat = w; w += (size_t)T_ * B_ * S_;        // 192000
    float* sum_g  = w; w += (size_t)B_ * S_;             // 3200

    // 1) embedding gather + constant decoder input projection
    embed_gather<<<B_ * S_, E_, 0, stream>>>(input, embedding, emb);
    precompute_dec_in<<<(4 * H2_ + 255) / 256, 256, 0, stream>>>(start_id, embedding, Wih_d, b_d, pre);

    // 2) encoder input projections (batched over all timesteps), WMMA f16
    dim3 g1((4 * H_ + 255) / 256, (B_ * S_ + 15) / 16);
    gemm_wmma_f16<<<g1, 256, 0, stream>>>(emb, E_, Wih_f, 4 * H_, b_f, Xf, 4 * H_, B_ * S_, 4 * H_, E_);
    gemm_wmma_f16<<<g1, 256, 0, stream>>>(emb, E_, Wih_b, 4 * H_, b_b, Xb, 4 * H_, B_ * S_, 4 * H_, E_);

    // 3) recurrent encoder scan (1 persistent block per direction)
    lstm_encoder<<<2, 1024, 0, stream>>>(Xf, Xb, Whh_f, Whh_b, h_enc, h0, c0);

    // 4) decoder scan (persistent block): produces hs_all, alphat, p_u
    decoder_scan<<<1, 1024, 0, stream>>>(input, pad_id, pre, Whh_d, W_e, W_d,
                                         h_enc, h0, c0, lin_u_w, lin_u_b,
                                         buf, hs_all, p_u, alphat, sum_g);

    // 5) hoisted vocab projection: (480 x 1536) @ (1536 x 50000) -> logits
    dim3 g2((V_ + 255) / 256, (T_ * B_ + 15) / 16);
    gemm_wmma_f16<<<g2, 256, 0, stream>>>(hs_all, HS3_, lin_out_w, V_, lin_out_b, out, V_, T_ * B_, V_, HS3_);

    // 6) softmax + p_u scale, then pointer scatter
    softmax_scale<<<T_ * B_, 256, 0, stream>>>(out, p_u);
    scatter_ptr<<<(T_ * B_ * S_ + 255) / 256, 256, 0, stream>>>(out, p_u, alphat, input);
}
